// AtomAttention_59682865545544
// MI455X (gfx1250) — matrix operand order repaired and checked by reading, not compiled, not run on hardware
//
#include <hip/hip_runtime.h>

#define B_ 16
#define N_ 1024
#define H_ 128

typedef __attribute__((ext_vector_type(16))) __bf16        v16bf;
typedef __attribute__((ext_vector_type(8)))  float         v8f;
typedef __attribute__((ext_vector_type(4)))  unsigned int  v4ui;
typedef __attribute__((ext_vector_type(8)))  int           v8si;
typedef __attribute__((ext_vector_type(4)))  int           v4si;

#define LOG2E_F  1.4426950408889634f
#define SCALE_F  0.08838834764831845f    /* 1/sqrt(128) */

static __device__ __forceinline__ v8f wmma_bf16(v16bf a, v16bf b, v8f c) {
    return __builtin_amdgcn_wmma_f32_16x16x32_bf16(
        false, a, false, b, (short)0, c, false, false);
}

#if __has_builtin(__builtin_amdgcn_tensor_load_to_lds)
#define HAVE_TDM 1
#if defined(__clang_major__) && __clang_major__ >= 23
#define TDM_LOAD(g0, g1, g2, g3) \
    __builtin_amdgcn_tensor_load_to_lds((g0), (g1), (g2), (g3), (v8si){0,0,0,0,0,0,0,0}, 0)
#else
#define TDM_LOAD(g0, g1, g2, g3) \
    __builtin_amdgcn_tensor_load_to_lds((g0), (g1), (g2), (g3), 0)
#endif
#else
#define HAVE_TDM 0
#endif

#if HAVE_TDM
// Issue a 2-D TDM descriptor: tile_d0 x tile_d1 elements (size 1<<ds_code B),
// global row stride = stride0 elements, packed per CDNA5 ISA 8.3/8.4
// (count=1, type=2; x runs fastest, written contiguously into LDS).
static __device__ __forceinline__ void tdm_load_2d(unsigned lds_addr, const void* gptr,
                                                   unsigned ds_code,
                                                   unsigned td0, unsigned td1,
                                                   unsigned tile0, unsigned tile1,
                                                   unsigned stride0)
{
    const unsigned long long ga = (unsigned long long)(size_t)gptr;
    v4ui g0 = { 1u,                                    // count=1 (valid descriptor)
                lds_addr,                              // LDS byte address
                (unsigned)ga,                          // global_addr[31:0]
                (unsigned)((ga >> 32) & 0x1FFFFFFu) | (2u << 30) }; // addr[56:32]|type=2
    v8si g1 = { (int)(ds_code << 16),                                     // data_size
                (int)((td0 & 0xFFFFu) << 16),                             // tensor_dim0 lo
                (int)(((td0 >> 16) & 0xFFFFu) | ((td1 & 0xFFFFu) << 16)),
                (int)(((td1 >> 16) & 0xFFFFu) | ((tile0 & 0xFFFFu) << 16)),
                (int)(tile1 & 0xFFFFu),                                   // tile_dim1
                (int)stride0, 0, 0 };                                     // dim0 stride
    v4si gz = {0, 0, 0, 0};
    TDM_LOAD(g0, g1, gz, gz);
}
#define SMEM_BYTES (4 * 13312)   /* per wave: 2 bufs x 3 x 2KB bias + 1KB pP */
#else
#define SMEM_BYTES (4 * 1024)    /* per wave: 1KB pP */
#endif

// ---------------------------------------------------------------------------
// Flash attention (placed first so the disasm snippet shows this kernel).
// Block = 4 independent waves = 4 query tiles of one batch; no barriers.
// Per 32-key iteration: 8 S-WMMAs (bf16, f32 acc) + online softmax (exp2,
// scale*log2e pre-folded into q) + 8 PV-WMMAs. The 192MB bias stream — the
// only HBM-bound traffic — is double-buffered per wave via the Tensor Data
// Mover: issue 3 tiles for iter+1, s_wait_tensorcnt(3) => iter's tiles ready.
// Grid: 256 blocks x 128 threads (16 blocks per batch).
// ---------------------------------------------------------------------------
__global__ __launch_bounds__(128)
void flash_kernel(const float* __restrict__ m,
                  const float* __restrict__ adj,
                  const float* __restrict__ dist,
                  const float* __restrict__ cou,
                  const __bf16* __restrict__ qkv,
                  float* __restrict__ out)
{
    __shared__ __align__(128) char smem[SMEM_BYTES];

    const int widx = threadIdx.x >> 5;
    const int lane = threadIdx.x & 31;
    const int half = lane >> 4;
    const int lr   = lane & 15;

    const int b_blk = blockIdx.x >> 4;                       // uniform per block
    const int qt    = ((blockIdx.x & 15) << 2) + widx;       // query tile 0..63

    const __bf16* qb = qkv;
    const __bf16* kb = qkv + (size_t)1 * B_ * N_ * H_;
    const __bf16* vt = qkv + (size_t)2 * B_ * N_ * H_;       // vT[B][H][N]

#if HAVE_TDM
    char*   wbase = smem + widx * 13312;                     // this wave's slice
    __bf16* pP    = (__bf16*)(wbase + 12288);                // 16x32 bf16
#else
    __bf16* pP    = (__bf16*)(smem + widx * 1024);
#endif

    // --- Q fragments (A layout), pre-scaled by scale*log2e ---
    v16bf aq[4];
    {
        const __bf16* qrow = qb + ((size_t)b_blk * N_ + qt * 16 + lr) * H_;
        #pragma unroll
        for (int kk = 0; kk < 4; ++kk) {
            const int ka = kk * 32 + half * 16;
            #pragma unroll
            for (int j = 0; j < 8; ++j) {
                aq[kk][j]     = qrow[ka + j];
                aq[kk][j + 8] = qrow[ka + 8 + j];
            }
        }
    }

    v8f O[8];
    #pragma unroll
    for (int h = 0; h < 8; ++h) O[h] = (v8f){0.f,0.f,0.f,0.f,0.f,0.f,0.f,0.f};
    float marr[8], larr[8];
    #pragma unroll
    for (int r = 0; r < 8; ++r) { marr[r] = -1e30f; larr[r] = 0.f; }

    const int NT = N_ / 32;

#if HAVE_TDM
    // Bias tile of this wave's 16 query rows x 32 keys, f32 (data_size code 2).
    const size_t grow0 = ((size_t)b_blk * N_ + qt * 16) * N_;
    {   // preload buffer 0 for kt2 = 0
        tdm_load_2d((unsigned)(size_t)(void*)(wbase + 0),    adj  + grow0, 2, 32, 16, 32, 16, N_);
        tdm_load_2d((unsigned)(size_t)(void*)(wbase + 2048), dist + grow0, 2, 32, 16, 32, 16, N_);
        tdm_load_2d((unsigned)(size_t)(void*)(wbase + 4096), cou  + grow0, 2, 32, 16, 32, 16, N_);
    }
#endif

    for (int kt2 = 0; kt2 < NT; ++kt2) {
#if HAVE_TDM
        const int cb = kt2 & 1;
        if (kt2 + 1 < NT) {
            // Issue next 3 tiles, then wait for the *previous* 3 (TENSORcnt
            // decrements in order): double-buffered async prefetch.
            char* nxt = wbase + (((kt2 + 1) & 1) ? 6144 : 0);
            const size_t g = grow0 + (size_t)(kt2 + 1) * 32;
            tdm_load_2d((unsigned)(size_t)(void*)(nxt + 0),    adj  + g, 2, 32, 16, 32, 16, N_);
            tdm_load_2d((unsigned)(size_t)(void*)(nxt + 2048), dist + g, 2, 32, 16, 32, 16, N_);
            tdm_load_2d((unsigned)(size_t)(void*)(nxt + 4096), cou  + g, 2, 32, 16, 32, 16, N_);
            __builtin_amdgcn_s_wait_tensorcnt(3);
        } else {
            __builtin_amdgcn_s_wait_tensorcnt(0);
        }
        const float* badj = (const float*)(wbase + cb * 6144);
        const float* bdst = (const float*)(wbase + cb * 6144 + 2048);
        const float* bcou = (const float*)(wbase + cb * 6144 + 4096);
#else
        if (kt2 + 1 < NT) {
            const size_t p = ((size_t)b_blk * N_ + qt * 16 + 8 * half) * N_
                           + (size_t)(kt2 + 1) * 32 + 2 * lr;
            __builtin_prefetch(adj + p);
            __builtin_prefetch(dist + p);
            __builtin_prefetch(cou + p);
        }
#endif

        // --- S = q'@k^T + bias*log2e (units of log2e), 2 x 16x16 tiles ---
        v8f S[2];
        #pragma unroll
        for (int sub = 0; sub < 2; ++sub) {
            v8f s = {0.f,0.f,0.f,0.f,0.f,0.f,0.f,0.f};
            const __bf16* krow = kb + ((size_t)b_blk * N_ + (size_t)kt2 * 32
                                       + sub * 16 + lr) * H_;   // B: col = key
            #pragma unroll
            for (int kk = 0; kk < 4; ++kk) {
                v16bf kf;
                const int kbase = kk * 32 + half * 16;
                #pragma unroll
                for (int j = 0; j < 16; ++j) kf[j] = krow[kbase + j];
                s = wmma_bf16(aq[kk], kf, s);
            }
            #pragma unroll
            for (int r = 0; r < 8; ++r) {
#if HAVE_TDM
                const int bi = (8 * half + r) * 32 + sub * 16 + lr;
                const float bsum = badj[bi] + bdst[bi] + bcou[bi];
#else
                const int row = qt * 16 + 8 * half + r;
                const size_t idx = ((size_t)b_blk * N_ + row) * N_
                                 + (size_t)kt2 * 32 + sub * 16 + lr;
                const float bsum = __builtin_nontemporal_load(adj + idx)
                                 + __builtin_nontemporal_load(dist + idx)
                                 + __builtin_nontemporal_load(cou + idx);
#endif
                s[r] = fmaf(bsum, LOG2E_F, s[r]);
            }
            S[sub] = s;
        }

        // --- Online softmax (stats replicated across each 16-lane half) ---
        #pragma unroll
        for (int r = 0; r < 8; ++r) {
            float v = fmaxf(S[0][r], S[1][r]);
            #pragma unroll
            for (int off = 1; off < 16; off <<= 1)
                v = fmaxf(v, __shfl_xor(v, off, 32));
            const float mnew  = fmaxf(marr[r], v);
            const float alpha = __builtin_amdgcn_exp2f(marr[r] - mnew);
            const float p0 = __builtin_amdgcn_exp2f(S[0][r] - mnew);
            const float p1 = __builtin_amdgcn_exp2f(S[1][r] - mnew);
            float rs = p0 + p1;
            #pragma unroll
            for (int off = 1; off < 16; off <<= 1)
                rs += __shfl_xor(rs, off, 32);
            larr[r] = larr[r] * alpha + rs;
            marr[r] = mnew;
            #pragma unroll
            for (int h = 0; h < 8; ++h) O[h][r] *= alpha;
            pP[(8 * half + r) * 32 + lr]      = (__bf16)p0;   // C->A transpose stash
            pP[(8 * half + r) * 32 + 16 + lr] = (__bf16)p1;
        }

        // DS ops are in-order per wave; wait, then reload P in A layout.
        // (This also retires the bias DS reads before the TDM overwrites
        //  their buffer at the top of the next iteration.)
        asm volatile("s_wait_dscnt 0" ::: "memory");
        v16bf pf;
        #pragma unroll
        for (int j = 0; j < 8; ++j) {
            pf[j]     = pP[lr * 32 + half * 16 + j];
            pf[j + 8] = pP[lr * 32 + half * 16 + 8 + j];
        }

        // --- O += P @ V: vT layout makes B-fragments 32B-contiguous ---
        #pragma unroll
        for (int h = 0; h < 8; ++h) {
            v16bf vf;
            const __bf16* vrow = vt + ((size_t)b_blk * H_ + h * 16 + lr) * N_
                               + (size_t)kt2 * 32 + half * 16;
            #pragma unroll
            for (int j = 0; j < 16; ++j) vf[j] = vrow[j];
            O[h] = wmma_bf16(pf, vf, O[h]);
        }
    }

    // --- Epilogue: normalize, add residual, store f32 ---
    #pragma unroll
    for (int h = 0; h < 8; ++h) {
        #pragma unroll
        for (int r = 0; r < 8; ++r) {
            const int row = qt * 16 + 8 * half + r;
            const size_t idx = ((size_t)b_blk * N_ + row) * H_ + h * 16 + lr;
            __builtin_nontemporal_store(O[h][r] / larr[r] + m[idx], out + idx);
        }
    }
}

// ---------------------------------------------------------------------------
// QKV projection (bf16 WMMA, f32 accumulate).
//   q -> pre-scaled by (1/sqrt(H))*log2(e), layout [B][N][H]
//   k -> layout [B][N][H]
//   v -> TRANSPOSED vT[B][H][N] so P@V B-fragments load contiguously
// ---------------------------------------------------------------------------
__global__ __launch_bounds__(256)
void qkv_kernel(const float* __restrict__ m,
                const float* __restrict__ Wq, const float* __restrict__ bq,
                const float* __restrict__ Wk, const float* __restrict__ bk,
                const float* __restrict__ Wv, const float* __restrict__ bv,
                __bf16* __restrict__ qkv)
{
    const int wid  = blockIdx.x * 8 + (threadIdx.x >> 5);
    const int lane = threadIdx.x & 31;
    const int half = lane >> 4;
    const int lr   = lane & 15;

    const int ht  = wid & 7;
    const int nt  = (wid >> 3) & 63;
    const int b   = (wid >> 9) & 15;
    const int mat = wid >> 13;                     // 0=q,1=k,2=v

    const float* W    = (mat == 0) ? Wq : (mat == 1 ? Wk : Wv);
    const float* bvec = (mat == 0) ? bq : (mat == 1 ? bk : bv);
    __bf16* out = qkv + (size_t)mat * B_ * N_ * H_;

    const int arow = nt * 16 + lr;                 // A: M = lane%16
    const int wrow = ht * 16 + lr;                 // B: N = lane%16 (W row = out col)
    const float* mrow = m + ((size_t)b * N_ + arow) * H_;
    const float* wr   = W + (size_t)wrow * H_;

    v8f c = {0.f,0.f,0.f,0.f,0.f,0.f,0.f,0.f};
    #pragma unroll
    for (int kk = 0; kk < 4; ++kk) {
        v16bf a, bb;
        const int ka = kk * 32 + half * 16;        // A: K = 16*half + {j, 8+j}
        const int kw = kk * 32 + half * 16;        // B: K = 16*half + j
        #pragma unroll
        for (int j = 0; j < 8; ++j) {
            a[j]     = (__bf16)mrow[ka + j];
            a[j + 8] = (__bf16)mrow[ka + 8 + j];
        }
        #pragma unroll
        for (int j = 0; j < 16; ++j) bb[j] = (__bf16)wr[kw + j];
        c = wmma_bf16(a, bb, c);
    }

    const float bias   = bvec[wrow];
    const float premul = (mat == 0) ? (SCALE_F * LOG2E_F) : 1.0f;
    #pragma unroll
    for (int r = 0; r < 8; ++r) {                  // C/D: row = 8*half + r, col = lr
        const int row   = nt * 16 + 8 * half + r;
        const float val = (c[r] + bias) * premul;
        if (mat == 2)  // vT[b][h=wrow][key=row]  (8 consecutive keys per lane)
            out[((size_t)b * H_ + wrow) * N_ + row] = (__bf16)val;
        else
            out[((size_t)b * N_ + row) * H_ + wrow] = (__bf16)val;
    }
}

// ---------------------------------------------------------------------------
extern "C" void kernel_launch(void* const* d_in, const int* in_sizes, int n_in,
                              void* d_out, int out_size, void* d_ws, size_t ws_size,
                              hipStream_t stream) {
    (void)in_sizes; (void)n_in; (void)out_size; (void)ws_size;
    const float* m    = (const float*)d_in[0];
    const float* adj  = (const float*)d_in[1];
    const float* dist = (const float*)d_in[2];
    const float* cou  = (const float*)d_in[3];
    const float* Wq   = (const float*)d_in[4];
    const float* bq   = (const float*)d_in[5];
    const float* Wk   = (const float*)d_in[6];
    const float* bk   = (const float*)d_in[7];
    const float* Wv   = (const float*)d_in[8];
    const float* bv   = (const float*)d_in[9];
    float*  out = (float*)d_out;
    __bf16* qkv = (__bf16*)d_ws;   // 3 * B*N*H bf16 = 12 MB scratch

    qkv_kernel<<<dim3(3072), dim3(256), 0, stream>>>(m, Wq, bq, Wk, bk, Wv, bv, qkv);
    flash_kernel<<<dim3(256), dim3(128), 0, stream>>>(m, adj, dist, cou, qkv, out);
}
